// CutsEmbedding_76828374991402
// MI455X (gfx1250) — compile-verified, matrix operand order checked by reading, not run on hardware
//
#include <hip/hip_runtime.h>
#include <hip/hip_bf16.h>

typedef __attribute__((ext_vector_type(16))) _Float16 v16h;
typedef __attribute__((ext_vector_type(8)))  _Float16 v8h;
typedef __attribute__((ext_vector_type(4)))  _Float16 v4h;
typedef __attribute__((ext_vector_type(8)))  float    v8f;

#define WAVES_PER_BLOCK 8
#define XCOLS 96   // max padded K for node kernel (fv: 80 -> 96)

struct Seg {
    const float* ptr;   // feature table, row stride = dim
    const int*   idx;   // gather indices (nullptr => identity)
    const float* cnt;   // per-row edge counts for mean (nullptr => none)
    int          dim;
};

// K index within a 32-wide K-chunk for element e (0..15) of a 16-bit WMMA
// A/B operand in a given lane (ISA 7.12.2 layout).
__device__ __forceinline__ int kmap(int e, int lane) {
    return (e & 7) + ((e & 8) << 1) + ((lane & 16) >> 1);
}

// Pre-swizzle a [Ktot x 32] f32 row-major weight matrix into the f16 B-operand
// layout: dst[((kc*2 + nt)*32 + lane)*16 + e]. Zero-pads K up to KC*32.
__global__ void pack_w16_kernel(const float* __restrict__ w, int Ktot, int KC,
                                _Float16* __restrict__ dst) {
    int t = blockIdx.x * blockDim.x + threadIdx.x;
    int total = KC * 2 * 32 * 16;
    if (t >= total) return;
    int e    = t & 15;
    int lane = (t >> 4) & 31;
    int nt   = (t >> 9) & 1;
    int kc   = t >> 10;
    int K = kc * 32 + kmap(e, lane);
    int N = nt * 16 + (lane & 15);
    float v = (K < Ktot) ? w[K * 32 + N] : 0.0f;
    dst[t] = (_Float16)v;
}

// ---------------------------------------------------------------------------
// Edge kernel: one wave32 per 16-edge tile. Inputs (3 gathered segments,
// total K <= 64) staged straight into LDS (f32) with async global->LDS DMA
// (ASYNCcnt path), then relu(X@W1+b1)@W2+b2 via f16 WMMA, scatter-add out.
// ---------------------------------------------------------------------------
__global__ __launch_bounds__(32 * WAVES_PER_BLOCK)
void edge_mlp_kernel(int n, Seg s0, Seg s1, Seg s2,
                     const _Float16* __restrict__ w1p, const float* __restrict__ b1,
                     const _Float16* __restrict__ w2p, const float* __restrict__ b2,
                     const int* __restrict__ outIdx,
                     float* __restrict__ accum, float* __restrict__ cntOut)
{
    __shared__ float    Xf[WAVES_PER_BLOCK][16][64];
    __shared__ _Float16 Hs[WAVES_PER_BLOCK][16][32];
    const int lane = threadIdx.x & 31;
    const int wv   = threadIdx.x >> 5;
    const int nTiles = (n + 15) >> 4;
    const int tile = blockIdx.x * WAVES_PER_BLOCK + wv;   // uniform per wave
    if (tile >= nTiles) return;
    const int base = tile << 4;
    const int r  = lane & 15;            // row (M) / output channel (N%16)
    const int kb = (lane & 16) >> 1;     // 0 or 8: K-half select

    // ---- async-stage the three segments into LDS as f32 (16B per lane/op)
    {
        const Seg* segs[3] = {&s0, &s1, &s2};
        int colBase = 0;
        #pragma unroll
        for (int sgi = 0; sgi < 3; ++sgi) {
            const Seg& s = *segs[sgi];
            const int prs = (s.dim == 16) ? 2 : 3;        // 16B chunks per row = 4 or 8
            const int nChunks = 16 << prs;
            for (int c = lane; c < nChunks; c += 32) {
                const int row = c >> prs;
                const int ch  = c & ((1 << prs) - 1);
                const int m = base + row;
                if (m < n) {
                    const long gi = s.idx ? (long)s.idx[m] : (long)m;
                    const float* gp = s.ptr + gi * s.dim + (ch << 2);
                    unsigned lp = (unsigned)(uintptr_t)&Xf[wv][row][colBase + (ch << 2)];
                    asm volatile("global_load_async_to_lds_b128 %0, %1, off"
                                 :: "v"(lp), "v"(gp) : "memory");
                }
            }
            colBase += s.dim;
        }
        // zero K-padding columns (e.g. K=48 padded to 64)
        if (colBase < 64) {
            v8f zz = {0.f,0.f,0.f,0.f,0.f,0.f,0.f,0.f};
            *(v8f*)&Xf[wv][r][colBase + (lane >> 4) * 8] = zz;   // covers 16 cols
        }
    }
    asm volatile("s_wait_asynccnt 0x0" ::: "memory");
    asm volatile("s_wait_dscnt 0x0" ::: "memory");

    // ---- layer 1: X[16 x 64] @ W1[64 x 32]  (KC = 2 fixed)
    v8f acc0 = {0.f,0.f,0.f,0.f,0.f,0.f,0.f,0.f};
    v8f acc1 = {0.f,0.f,0.f,0.f,0.f,0.f,0.f,0.f};
    const v16h* w1v = (const v16h*)w1p;
    #pragma unroll
    for (int kc = 0; kc < 2; ++kc) {
        const v8f lo = *(const v8f*)&Xf[wv][r][kc * 32 + kb];
        const v8f hi = *(const v8f*)&Xf[wv][r][kc * 32 + kb + 16];
        v16h a;
        #pragma unroll
        for (int i = 0; i < 8; ++i) { a[i] = (_Float16)lo[i]; a[i + 8] = (_Float16)hi[i]; }
        const v16h bA = w1v[(kc * 2 + 0) * 32 + lane];
        const v16h bB = w1v[(kc * 2 + 1) * 32 + lane];
        acc0 = __builtin_amdgcn_wmma_f32_16x16x32_f16(false, a, false, bA, (short)0, acc0, false, false);
        acc1 = __builtin_amdgcn_wmma_f32_16x16x32_f16(false, a, false, bB, (short)0, acc1, false, false);
    }

    // ---- bias + ReLU, C-layout -> A-layout rearrange through LDS
    {
        const float bb0 = b1[r];
        const float bb1 = b1[16 + r];
        #pragma unroll
        for (int q = 0; q < 8; ++q) {
            Hs[wv][q + kb][r]      = (_Float16)fmaxf(acc0[q] + bb0, 0.f);
            Hs[wv][q + kb][16 + r] = (_Float16)fmaxf(acc1[q] + bb1, 0.f);
        }
    }
    asm volatile("s_wait_dscnt 0x0" ::: "memory");

    // ---- layer 2: H[16 x 32] @ W2[32 x 32]
    v16h a2;
    {
        const v8h lo = *(const v8h*)&Hs[wv][r][kb];
        const v8h hi = *(const v8h*)&Hs[wv][r][kb + 16];
        #pragma unroll
        for (int i = 0; i < 8; ++i) { a2[i] = lo[i]; a2[i + 8] = hi[i]; }
    }
    const v16h* w2v = (const v16h*)w2p;
    const v16h b20 = w2v[lane];
    const v16h b21 = w2v[32 + lane];
    v8f z = {0.f,0.f,0.f,0.f,0.f,0.f,0.f,0.f};
    v8f d0 = __builtin_amdgcn_wmma_f32_16x16x32_f16(false, a2, false, b20, (short)0, z, false, false);
    v8f d1 = __builtin_amdgcn_wmma_f32_16x16x32_f16(false, a2, false, b21, (short)0, z, false, false);

    // ---- scatter-add into accumulators (+ per-edge count)
    const float c20 = b2[r];
    const float c21 = b2[16 + r];
    #pragma unroll
    for (int q = 0; q < 8; ++q) {
        const int m = base + q + kb;
        if (m < n) {
            const long nd = outIdx[m];
            atomicAdd(accum + nd * 32 + r,      d0[q] + c20);
            atomicAdd(accum + nd * 32 + 16 + r, d1[q] + c21);
        }
    }
    if (lane < 16) {
        const int m = base + lane;
        if (m < n) atomicAdd(cntOut + outIdx[m], 1.0f);
    }
}

// ---------------------------------------------------------------------------
// Node kernel: generic staging (supports mean scaling by counts), dense out.
// ---------------------------------------------------------------------------
__global__ __launch_bounds__(32 * WAVES_PER_BLOCK)
void node_mlp_kernel(int n, Seg s0, Seg s1, Seg s2, int KC,
                     const _Float16* __restrict__ w1p, const float* __restrict__ b1,
                     const _Float16* __restrict__ w2p, const float* __restrict__ b2,
                     float* __restrict__ outDense)
{
    __shared__ _Float16 Xs[WAVES_PER_BLOCK][16][XCOLS];
    __shared__ _Float16 Hs[WAVES_PER_BLOCK][16][32];
    const int lane = threadIdx.x & 31;
    const int wv   = threadIdx.x >> 5;
    const int nTiles = (n + 15) >> 4;
    const int tile = blockIdx.x * WAVES_PER_BLOCK + wv;
    if (tile >= nTiles) return;
    const int base = tile << 4;
    const int r  = lane & 15;
    const int kb = (lane & 16) >> 1;
    const int Kpad = KC * 32;

    // ---- stage input tile into LDS as f16; lanes 0-15 low half cols, 16-31 high
    {
        const int m = base + r;
        const bool valid = (m < n);
        const int half = Kpad >> 1;
        const int cbeg = (lane >> 4) * half;
        const int cend = cbeg + half;
        const int d0 = s0.dim, d01 = d0 + s1.dim, d012 = d01 + s2.dim;
        long i0 = 0, i1 = 0, i2 = 0;
        float inv0 = 1.f, inv1 = 1.f, inv2 = 1.f;
        if (valid) {
            i0 = s0.idx ? (long)s0.idx[m] : (long)m;
            i1 = s1.idx ? (long)s1.idx[m] : (long)m;
            i2 = (s2.ptr && s2.idx) ? (long)s2.idx[m] : (long)m;
            if (s0.cnt) inv0 = 1.f / fmaxf(s0.cnt[i0], 1.f);
            if (s1.cnt) inv1 = 1.f / fmaxf(s1.cnt[i1], 1.f);
            if (s2.ptr && s2.cnt) inv2 = 1.f / fmaxf(s2.cnt[i2], 1.f);
        }
        for (int c = cbeg; c < cend; c += 4) {
            float4 v = make_float4(0.f, 0.f, 0.f, 0.f);
            if (valid) {
                if (c < d0) {
                    float4 t4 = *(const float4*)(s0.ptr + i0 * s0.dim + c);
                    v = make_float4(t4.x*inv0, t4.y*inv0, t4.z*inv0, t4.w*inv0);
                } else if (c < d01) {
                    float4 t4 = *(const float4*)(s1.ptr + i1 * s1.dim + (c - d0));
                    v = make_float4(t4.x*inv1, t4.y*inv1, t4.z*inv1, t4.w*inv1);
                } else if (s2.ptr && c < d012) {
                    float4 t4 = *(const float4*)(s2.ptr + i2 * s2.dim + (c - d01));
                    v = make_float4(t4.x*inv2, t4.y*inv2, t4.z*inv2, t4.w*inv2);
                }
            }
            v4h hv4;
            hv4[0] = (_Float16)v.x; hv4[1] = (_Float16)v.y;
            hv4[2] = (_Float16)v.z; hv4[3] = (_Float16)v.w;
            *(v4h*)&Xs[wv][r][c] = hv4;
        }
    }
    asm volatile("s_wait_dscnt 0x0" ::: "memory");

    // ---- layer 1
    v8f acc0 = {0.f,0.f,0.f,0.f,0.f,0.f,0.f,0.f};
    v8f acc1 = {0.f,0.f,0.f,0.f,0.f,0.f,0.f,0.f};
    const v16h* w1v = (const v16h*)w1p;
    for (int kc = 0; kc < KC; ++kc) {
        v16h a;
        const v8h lo = *(const v8h*)&Xs[wv][r][kc * 32 + kb];
        const v8h hi = *(const v8h*)&Xs[wv][r][kc * 32 + kb + 16];
        #pragma unroll
        for (int i = 0; i < 8; ++i) { a[i] = lo[i]; a[i + 8] = hi[i]; }
        const v16h bA = w1v[(kc * 2 + 0) * 32 + lane];
        const v16h bB = w1v[(kc * 2 + 1) * 32 + lane];
        acc0 = __builtin_amdgcn_wmma_f32_16x16x32_f16(false, a, false, bA, (short)0, acc0, false, false);
        acc1 = __builtin_amdgcn_wmma_f32_16x16x32_f16(false, a, false, bB, (short)0, acc1, false, false);
    }

    // ---- bias + ReLU -> Hs
    {
        const float bb0 = b1[r];
        const float bb1 = b1[16 + r];
        #pragma unroll
        for (int q = 0; q < 8; ++q) {
            Hs[wv][q + kb][r]      = (_Float16)fmaxf(acc0[q] + bb0, 0.f);
            Hs[wv][q + kb][16 + r] = (_Float16)fmaxf(acc1[q] + bb1, 0.f);
        }
    }
    asm volatile("s_wait_dscnt 0x0" ::: "memory");

    // ---- layer 2
    v16h a2;
    {
        const v8h lo = *(const v8h*)&Hs[wv][r][kb];
        const v8h hi = *(const v8h*)&Hs[wv][r][kb + 16];
        #pragma unroll
        for (int i = 0; i < 8; ++i) { a2[i] = lo[i]; a2[i + 8] = hi[i]; }
    }
    const v16h* w2v = (const v16h*)w2p;
    const v16h b20 = w2v[lane];
    const v16h b21 = w2v[32 + lane];
    v8f z = {0.f,0.f,0.f,0.f,0.f,0.f,0.f,0.f};
    v8f d0 = __builtin_amdgcn_wmma_f32_16x16x32_f16(false, a2, false, b20, (short)0, z, false, false);
    v8f d1 = __builtin_amdgcn_wmma_f32_16x16x32_f16(false, a2, false, b21, (short)0, z, false, false);

    const float c20 = b2[r];
    const float c21 = b2[16 + r];
    #pragma unroll
    for (int q = 0; q < 8; ++q) {
        const long m = base + q + kb;
        if (m < n) {
            outDense[m * 32 + r]      = d0[q] + c20;
            outDense[m * 32 + 16 + r] = d1[q] + c21;
        }
    }
}

extern "C" void kernel_launch(void* const* d_in, const int* in_sizes, int n_in,
                              void* d_out, int out_size, void* d_ws, size_t ws_size,
                              hipStream_t stream)
{
    (void)n_in; (void)out_size; (void)ws_size;
    const float* x_vars  = (const float*)d_in[0];
    const float* x_cons  = (const float*)d_in[1];
    const float* x_cuts  = (const float*)d_in[2];
    const float* ea_cons = (const float*)d_in[3];
    const float* ea_cuts = (const float*)d_in[4];
    const int* cons_row = (const int*)d_in[5];
    const int* cons_col = (const int*)d_in[6];
    const int* cuts_row = (const int*)d_in[7];
    const int* cuts_col = (const int*)d_in[8];
    const float* gv_w1 = (const float*)d_in[9];  const float* gv_b1 = (const float*)d_in[10];
    const float* gv_w2 = (const float*)d_in[11]; const float* gv_b2 = (const float*)d_in[12];
    const float* hv_w1 = (const float*)d_in[13]; const float* hv_b1 = (const float*)d_in[14];
    const float* hv_w2 = (const float*)d_in[15]; const float* hv_b2 = (const float*)d_in[16];
    const float* fv_w1 = (const float*)d_in[17]; const float* fv_b1 = (const float*)d_in[18];
    const float* fv_w2 = (const float*)d_in[19]; const float* fv_b2 = (const float*)d_in[20];
    const float* gc_w1 = (const float*)d_in[21]; const float* gc_b1 = (const float*)d_in[22];
    const float* gc_w2 = (const float*)d_in[23]; const float* gc_b2 = (const float*)d_in[24];
    const float* fc_w1 = (const float*)d_in[25]; const float* fc_b1 = (const float*)d_in[26];
    const float* fc_w2 = (const float*)d_in[27]; const float* fc_b2 = (const float*)d_in[28];

    const int NV = in_sizes[0] / 16;
    const int NU = in_sizes[2] / 16;
    const int EC = in_sizes[5];
    const int EU = in_sizes[7];

    // ---- workspace carve-up (256B aligned regions)
    char* ws = (char*)d_ws;
    size_t off = 0;
    auto take = [&](size_t bytes) -> char* {
        char* p = ws + off;
        off = (off + bytes + 255) & ~(size_t)255;
        return p;
    };
    float* aggr_g  = (float*)take((size_t)NV * 32 * sizeof(float));
    float* cnt_g   = (float*)take((size_t)NV * sizeof(float));
    float* aggr_h  = (float*)take((size_t)NV * 32 * sizeof(float));
    float* cnt_h   = (float*)take((size_t)NV * sizeof(float));
    float* aggr_gc = (float*)take((size_t)NU * 32 * sizeof(float));
    float* cnt_gc  = (float*)take((size_t)NU * sizeof(float));
    const size_t zeroBytes = off;                    // everything above must start at 0
    float* fv_out  = (float*)take((size_t)NV * 32 * sizeof(float));
    _Float16* gv_w1p = (_Float16*)take(2 * 1024 * sizeof(_Float16));
    _Float16* gv_w2p = (_Float16*)take(1 * 1024 * sizeof(_Float16));
    _Float16* hv_w1p = (_Float16*)take(2 * 1024 * sizeof(_Float16));
    _Float16* hv_w2p = (_Float16*)take(1 * 1024 * sizeof(_Float16));
    _Float16* fv_w1p = (_Float16*)take(3 * 1024 * sizeof(_Float16));
    _Float16* fv_w2p = (_Float16*)take(1 * 1024 * sizeof(_Float16));
    _Float16* gc_w1p = (_Float16*)take(2 * 1024 * sizeof(_Float16));
    _Float16* gc_w2p = (_Float16*)take(1 * 1024 * sizeof(_Float16));
    _Float16* fc_w1p = (_Float16*)take(2 * 1024 * sizeof(_Float16));
    _Float16* fc_w2p = (_Float16*)take(1 * 1024 * sizeof(_Float16));

    hipMemsetAsync(d_ws, 0, zeroBytes, stream);

    auto pack = [&](const float* w, int Ktot, int KC, _Float16* dst) {
        const int total = KC * 1024;
        pack_w16_kernel<<<(total + 255) / 256, 256, 0, stream>>>(w, Ktot, KC, dst);
    };
    pack(gv_w1, 48, 2, gv_w1p); pack(gv_w2, 32, 1, gv_w2p);
    pack(hv_w1, 48, 2, hv_w1p); pack(hv_w2, 32, 1, hv_w2p);
    pack(fv_w1, 80, 3, fv_w1p); pack(fv_w2, 32, 1, fv_w2p);
    pack(gc_w1, 64, 2, gc_w1p); pack(gc_w2, 32, 1, gc_w2p);
    pack(fc_w1, 48, 2, fc_w1p); pack(fc_w2, 32, 1, fc_w2p);

    auto blocks = [](int n) { return ((n + 15) / 16 + WAVES_PER_BLOCK - 1) / WAVES_PER_BLOCK; };

    // 1) gv over cons edges -> scatter into aggr_g by cons_col
    {
        Seg s0{x_vars, cons_col, nullptr, 16};
        Seg s1{x_cons, cons_row, nullptr, 16};
        Seg s2{ea_cons, nullptr, nullptr, 16};
        edge_mlp_kernel<<<blocks(EC), 32 * WAVES_PER_BLOCK, 0, stream>>>(
            EC, s0, s1, s2, gv_w1p, gv_b1, gv_w2p, gv_b2,
            cons_col, aggr_g, cnt_g);
    }
    // 2) hv over cuts edges -> scatter into aggr_h by cuts_col
    {
        Seg s0{x_vars, cuts_col, nullptr, 16};
        Seg s1{x_cuts, cuts_row, nullptr, 16};
        Seg s2{ea_cuts, nullptr, nullptr, 16};
        edge_mlp_kernel<<<blocks(EU), 32 * WAVES_PER_BLOCK, 0, stream>>>(
            EU, s0, s1, s2, hv_w1p, hv_b1, hv_w2p, hv_b2,
            cuts_col, aggr_h, cnt_h);
    }
    // 3) fv over variable nodes (means of aggr_g/aggr_h) -> fv_out
    {
        Seg s0{x_vars, nullptr, nullptr, 16};
        Seg s1{aggr_g, nullptr, cnt_g, 32};
        Seg s2{aggr_h, nullptr, cnt_h, 32};
        node_mlp_kernel<<<blocks(NV), 32 * WAVES_PER_BLOCK, 0, stream>>>(
            NV, s0, s1, s2, 3, fv_w1p, fv_b1, fv_w2p, fv_b2, fv_out);
    }
    // 4) gc over cuts edges (uses fv_out) -> scatter into aggr_gc by cuts_row
    {
        Seg s0{x_cuts, cuts_row, nullptr, 16};
        Seg s1{fv_out, cuts_col, nullptr, 32};
        Seg s2{ea_cuts, nullptr, nullptr, 16};
        edge_mlp_kernel<<<blocks(EU), 32 * WAVES_PER_BLOCK, 0, stream>>>(
            EU, s0, s1, s2, gc_w1p, gc_b1, gc_w2p, gc_b2,
            cuts_row, aggr_gc, cnt_gc);
    }
    // 5) fc over cut nodes -> d_out [NU, 32]
    {
        Seg s0{x_cuts, nullptr, nullptr, 16};
        Seg s1{aggr_gc, nullptr, cnt_gc, 32};
        Seg s2{nullptr, nullptr, nullptr, 0};
        node_mlp_kernel<<<blocks(NU), 32 * WAVES_PER_BLOCK, 0, stream>>>(
            NU, s0, s1, s2, 2, fc_w1p, fc_b1, fc_w2p, fc_b2, (float*)d_out);
    }
}
